// multiheadAttention_48086453846557
// MI455X (gfx1250) — matrix operand order repaired and checked by reading, not compiled, not run on hardware
//
#include <hip/hip_runtime.h>

typedef __bf16 bf16_t;
typedef float  v8f   __attribute__((ext_vector_type(8)));
typedef __bf16 v16bf __attribute__((ext_vector_type(16)));
typedef __bf16 v8bf  __attribute__((ext_vector_type(8)));
typedef __bf16 v4bf  __attribute__((ext_vector_type(4)));

#define DEV __device__ __forceinline__

constexpr int B = 4, S = 2048, DM = 1024, H = 16, DK = 64, DV = 64;

DEV int lane_id() { return threadIdx.x & 31; }

DEV v8f wmma_bf16(v16bf a, v16bf b, v8f c) {
  return __builtin_amdgcn_wmma_f32_16x16x32_bf16(
      /*neg_a=*/false, a, /*neg_b=*/false, b,
      /*c_mod=*/(short)0, c, /*reuse_a=*/false, /*reuse_b=*/false);
}

// A fragment: 16x32 bf16 tile, row-major with leading dim `ld` (elements).
// Lanes 0-15: row=lane, kappa 0..15; lanes 16-31: row=lane-16, kappa 16..31.
DEV v16bf load_a_frag(const bf16_t* base, int ld) {
  int lane = lane_id();
  const bf16_t* p = base + (lane & 15) * ld + (lane >> 4) * 16;
  v8bf lo = *(const v8bf*)p;         // kappa 0..7 (or 16..23)
  v8bf hi = *(const v8bf*)(p + 8);   // kappa 8..15 (or 24..31)
  v16bf r;
#pragma unroll
  for (int i = 0; i < 8; ++i) { r[i] = lo[i]; r[8 + i] = hi[i]; }
  return r;
}

// B fragment: 32x16, stored "column-contiguous": element B[kappa][n] at base[n*ld + kappa].
// Lane n holds column n, kappa 0..15 (lanes 0-15) / 16..31 (lanes 16-31), contiguous 32B.
DEV v16bf load_b_frag(const bf16_t* base, int ld) {
  int lane = lane_id();
  return *(const v16bf*)(base + (lane & 15) * ld + (lane >> 4) * 16);
}

// Issue one per-lane async global->LDS 16-byte copy (ASYNCcnt-tracked).
DEV void async_g2l_b128(unsigned lds_byte_addr, const void* gaddr) {
  asm volatile("global_load_async_to_lds_b128 %0, %1, off"
               :: "v"(lds_byte_addr),
                  "v"((unsigned long long)(uintptr_t)gaddr)
               : "memory");
}

// ---------------- format conversion kernels ----------------

__global__ void cvt_f32_bf16_kernel(const float* __restrict__ src,
                                    bf16_t* __restrict__ dst, int n) {
  int i = (blockIdx.x * blockDim.x + threadIdx.x) * 4;
  if (i >= n) return;
  float4 v = *(const float4*)(src + i);
  v4bf o;
  o[0] = (bf16_t)v.x; o[1] = (bf16_t)v.y; o[2] = (bf16_t)v.z; o[3] = (bf16_t)v.w;
  *(v4bf*)(dst + i) = o;
}

// src: [Hn, R, C] f32  ->  dst: [Hn, C, R] bf16
__global__ void transpose_cvt_kernel(const float* __restrict__ src,
                                     bf16_t* __restrict__ dst,
                                     int Hn, int R, int C) {
  long i = (long)blockIdx.x * blockDim.x + threadIdx.x;
  long total = (long)Hn * R * C;
  if (i >= total) return;
  int c = (int)(i % C);
  long t = i / C;
  int r = (int)(t % R);
  int hh = (int)(t / R);
  dst[((long)hh * C + c) * R + r] = (bf16_t)src[i];
}

// ---------------- QKV projection: X[B*S,DM] @ W[h][DM,DK] ----------------
// grid: (B*S/64, H, 3); block: 128 (4 waves, each a 32x32 quadrant of a 64x64 tile)
__global__ void __launch_bounds__(128)
qkv_proj_kernel(const bf16_t* __restrict__ qb, const bf16_t* __restrict__ kb,
                const bf16_t* __restrict__ vb, const bf16_t* __restrict__ wqt,
                const bf16_t* __restrict__ wkt, const bf16_t* __restrict__ wvt,
                bf16_t* __restrict__ Qp, bf16_t* __restrict__ Kp,
                bf16_t* __restrict__ Vt) {
  int z = blockIdx.z, h = blockIdx.y;
  const bf16_t* X = (z == 0) ? qb : (z == 1) ? kb : vb;
  const bf16_t* W = (z == 0) ? wqt : (z == 1) ? wkt : wvt;  // [H, DK, DM]

  int wave = threadIdx.x >> 5;
  int wm = wave >> 1, wn = wave & 1;
  int row0 = blockIdx.x * 64 + wm * 32;
  int col0 = wn * 32;

  v8f acc[2][2] = {};
  const bf16_t* xbase = X + (long)row0 * DM;
  const bf16_t* wbase = W + ((long)h * DK + col0) * DM;

  for (int d = 0; d < DM; d += 32) {
    v16bf a0 = load_a_frag(xbase + d, DM);
    v16bf a1 = load_a_frag(xbase + 16 * DM + d, DM);
    v16bf b0 = load_b_frag(wbase + d, DM);
    v16bf b1 = load_b_frag(wbase + 16 * DM + d, DM);
    acc[0][0] = wmma_bf16(a0, b0, acc[0][0]);
    acc[0][1] = wmma_bf16(a0, b1, acc[0][1]);
    acc[1][0] = wmma_bf16(a1, b0, acc[1][0]);
    acc[1][1] = wmma_bf16(a1, b1, acc[1][1]);
  }

  int lane = lane_id(), colc = lane & 15, hi = lane >> 4;
  float scale = (z == 0) ? 0.125f : 1.0f;  // fold 1/sqrt(DK) into Q
#pragma unroll
  for (int i = 0; i < 2; ++i)
#pragma unroll
    for (int j = 0; j < 2; ++j)
#pragma unroll
      for (int r = 0; r < 8; ++r) {
        int grow = row0 + i * 16 + r + 8 * hi;
        int col = col0 + j * 16 + colc;
        int b = grow >> 11, s = grow & (S - 1);
        float val = acc[i][j][r] * scale;
        if (z == 0)
          Qp[(((long)b * H + h) * S + s) * DK + col] = (bf16_t)val;
        else if (z == 1)
          Kp[(((long)b * H + h) * S + s) * DK + col] = (bf16_t)val;
        else  // store V transposed: [B,H,DV,S] so P@V B-fragments are contiguous
          Vt[(((long)b * H + h) * DV + col) * S + s] = (bf16_t)val;
      }
}

// ---------------- flash attention: 4-wave block owns 64 query rows ----------------
// K/V chunks staged once per block into LDS via async global->LDS copies, so the
// full K/V stream is read B*H*(S/64) times instead of B*H*(S/16) times (4x less).
// grid: B*H*(S/64); block: 128
__global__ void __launch_bounds__(128)
attn_kernel(const bf16_t* __restrict__ Qp, const bf16_t* __restrict__ Kp,
            const bf16_t* __restrict__ Vt, bf16_t* __restrict__ multi) {
  __shared__ bf16_t kLds[32 * 64];      // [key][kappa]  4KB
  __shared__ bf16_t vLds[64 * 32];      // [v][tau]      4KB
  __shared__ bf16_t pLds[4][16 * 32];   // per-wave P transpose 4KB

  int wave = threadIdx.x >> 5;
  int bh = blockIdx.x >> 5;            // b*H + h   (S/64 == 32)
  int rowblk = blockIdx.x & 31;
  int s0 = rowblk * 64 + wave * 16;
  int lane = lane_id(), colc = lane & 15, hi = lane >> 4;

  const bf16_t* Qb = Qp + ((long)bh * S + s0) * DK;
  const bf16_t* Kb = Kp + (long)bh * S * DK;
  const bf16_t* Vb = Vt + (long)bh * DV * S;

  unsigned kbase = (unsigned)(uintptr_t)(void*)kLds;
  unsigned vbase = (unsigned)(uintptr_t)(void*)vLds;

  v16bf aq0 = load_a_frag(Qb, DK);       // kappa 0..31 of Q rows
  v16bf aq1 = load_a_frag(Qb + 32, DK);  // kappa 32..63

  v8f o[4] = {};
  float mrow[8], lrow[8];
#pragma unroll
  for (int r = 0; r < 8; ++r) { mrow[r] = -1e30f; lrow[r] = 0.f; }

  bf16_t* pl = pLds[wave];
  for (int t = 0; t < S; t += 32) {
    // ---- stage K[t..t+31][0..64) (contiguous 4KB) and V[0..64)[t..t+32) ----
#pragma unroll
    for (int i = 0; i < 2; ++i) {
      int e = threadIdx.x + 128 * i;            // 256 x 16B transfers each
      async_g2l_b128(kbase + e * 16, Kb + (long)t * DK + e * 8);
      int vrow = e >> 2, vcol = (e & 3) * 8;
      async_g2l_b128(vbase + e * 16, Vb + (long)vrow * S + t + vcol);
    }
    asm volatile("s_wait_asynccnt 0" ::: "memory");
    __syncthreads();  // LDS chunk visible to all 4 waves

    // ---- scores: two 16x16 C tiles (keys t..t+15, t+16..t+31) ----
    v16bf bk0a = load_b_frag(kLds, 64);
    v16bf bk0b = load_b_frag(kLds + 32, 64);
    v16bf bk1a = load_b_frag(kLds + 16 * 64, 64);
    v16bf bk1b = load_b_frag(kLds + 16 * 64 + 32, 64);
    v8f c0 = {}, c1 = {};
    c0 = wmma_bf16(aq0, bk0a, c0);
    c0 = wmma_bf16(aq1, bk0b, c0);
    c1 = wmma_bf16(aq0, bk1a, c1);
    c1 = wmma_bf16(aq1, bk1b, c1);

    // ---- online softmax: row max over 32 cols (16-lane groups hold one row) ----
    float cand[8];
#pragma unroll
    for (int r = 0; r < 8; ++r) cand[r] = fmaxf(c0[r], c1[r]);
#pragma unroll
    for (int off = 1; off < 16; off <<= 1)
#pragma unroll
      for (int r = 0; r < 8; ++r)
        cand[r] = fmaxf(cand[r], __shfl_xor(cand[r], off, 32));

    float alpha[8];
#pragma unroll
    for (int r = 0; r < 8; ++r) {
      float mn = fmaxf(mrow[r], cand[r]);
      alpha[r] = __expf(mrow[r] - mn);
      mrow[r] = mn;
    }
#pragma unroll
    for (int r = 0; r < 8; ++r) {
      c0[r] = __expf(c0[r] - mrow[r]);
      c1[r] = __expf(c1[r] - mrow[r]);
      lrow[r] = lrow[r] * alpha[r] + c0[r] + c1[r];
    }
#pragma unroll
    for (int j = 0; j < 4; ++j)
#pragma unroll
      for (int r = 0; r < 8; ++r) o[j][r] *= alpha[r];

    // ---- C-layout -> A-layout transpose of P through per-wave LDS slice ----
#pragma unroll
    for (int r = 0; r < 8; ++r) {
      int row = r + 8 * hi;
      pl[row * 32 + colc] = (bf16_t)c0[r];
      pl[row * 32 + 16 + colc] = (bf16_t)c1[r];
    }
    asm volatile("s_wait_dscnt 0" ::: "memory");
    v16bf pA = load_a_frag(pl, 32);

    // ---- o += P(16x32) @ V(32x16) for 4 v-tiles, B-fragments from LDS ----
#pragma unroll
    for (int j = 0; j < 4; ++j) {
      v16bf bv = load_b_frag(vLds + (j * 16) * 32, 32);
      o[j] = wmma_bf16(pA, bv, o[j]);
    }
    __syncthreads();  // protect kLds/vLds from next iteration's staging
  }

  float inv[8];
#pragma unroll
  for (int r = 0; r < 8; ++r) {
    float sum = lrow[r];
#pragma unroll
    for (int off = 1; off < 16; off <<= 1) sum += __shfl_xor(sum, off, 32);
    inv[r] = 1.0f / sum;
  }

  int b = bh >> 4, h = bh & 15;
#pragma unroll
  for (int j = 0; j < 4; ++j)
#pragma unroll
    for (int r = 0; r < 8; ++r) {
      int srow = s0 + r + 8 * hi;
      multi[((long)(b * S + srow)) * DM + h * DV + j * 16 + colc] =
          (bf16_t)(o[j][r] * inv[r]);
    }
}

// ---------------- output projection: multi[B*S,DM] @ Wo[DM,DM] -> f32 ----------------
// grid: (B*S/64, DM/64); block: 128
__global__ void __launch_bounds__(128)
out_proj_kernel(const bf16_t* __restrict__ multi, const bf16_t* __restrict__ wot,
                float* __restrict__ out) {
  int wave = threadIdx.x >> 5;
  int wm = wave >> 1, wn = wave & 1;
  int row0 = blockIdx.x * 64 + wm * 32;
  int col0 = blockIdx.y * 64 + wn * 32;

  v8f acc[2][2] = {};
  const bf16_t* abase = multi + (long)row0 * DM;
  const bf16_t* bbase = wot + (long)col0 * DM;  // wot = Wo transposed [n][k]

  for (int d = 0; d < DM; d += 32) {
    v16bf a0 = load_a_frag(abase + d, DM);
    v16bf a1 = load_a_frag(abase + 16 * DM + d, DM);
    v16bf b0 = load_b_frag(bbase + d, DM);
    v16bf b1 = load_b_frag(bbase + 16 * DM + d, DM);
    acc[0][0] = wmma_bf16(a0, b0, acc[0][0]);
    acc[0][1] = wmma_bf16(a0, b1, acc[0][1]);
    acc[1][0] = wmma_bf16(a1, b0, acc[1][0]);
    acc[1][1] = wmma_bf16(a1, b1, acc[1][1]);
  }

  int lane = lane_id(), colc = lane & 15, hi = lane >> 4;
#pragma unroll
  for (int i = 0; i < 2; ++i)
#pragma unroll
    for (int j = 0; j < 2; ++j)
#pragma unroll
      for (int r = 0; r < 8; ++r)
        out[((long)(row0 + i * 16 + r + 8 * hi)) * DM + col0 + j * 16 + colc] =
            acc[i][j][r];
}

// ---------------- host launcher ----------------

extern "C" void kernel_launch(void* const* d_in, const int* in_sizes, int n_in,
                              void* d_out, int out_size, void* d_ws, size_t ws_size,
                              hipStream_t stream) {
  (void)in_sizes; (void)n_in; (void)out_size; (void)ws_size;
  const float* q  = (const float*)d_in[0];
  const float* k  = (const float*)d_in[1];
  const float* v  = (const float*)d_in[2];
  const float* Wq = (const float*)d_in[3];
  const float* Wk = (const float*)d_in[4];
  const float* Wv = (const float*)d_in[5];
  const float* Wo = (const float*)d_in[6];

  char* ws = (char*)d_ws;
  size_t off = 0;
  auto alloc = [&](size_t bytes) -> void* {
    void* p = ws + off;
    off += (bytes + 255) & ~(size_t)255;
    return p;
  };

  const size_t nTok = (size_t)B * S * DM;      // 8,388,608
  const size_t nW   = (size_t)H * DM * DK;     // 1,048,576
  const size_t nP   = (size_t)B * H * S * DK;  // 8,388,608

  bf16_t* qb  = (bf16_t*)alloc(nTok * 2);
  bf16_t* kb  = (bf16_t*)alloc(nTok * 2);
  bf16_t* vb  = (bf16_t*)alloc(nTok * 2);
  bf16_t* wqt = (bf16_t*)alloc(nW * 2);
  bf16_t* wkt = (bf16_t*)alloc(nW * 2);
  bf16_t* wvt = (bf16_t*)alloc(nW * 2);
  bf16_t* wot = (bf16_t*)alloc((size_t)DM * DM * 2);
  bf16_t* Qp  = (bf16_t*)alloc(nP * 2);
  bf16_t* Kp  = (bf16_t*)alloc(nP * 2);
  bf16_t* Vt  = (bf16_t*)alloc(nP * 2);
  bf16_t* mlt = (bf16_t*)alloc(nTok * 2);

  int cvtBlocks = (int)(nTok / 4 / 256);
  cvt_f32_bf16_kernel<<<cvtBlocks, 256, 0, stream>>>(q, qb, (int)nTok);
  cvt_f32_bf16_kernel<<<cvtBlocks, 256, 0, stream>>>(k, kb, (int)nTok);
  cvt_f32_bf16_kernel<<<cvtBlocks, 256, 0, stream>>>(v, vb, (int)nTok);
  transpose_cvt_kernel<<<(int)(nW / 256), 256, 0, stream>>>(Wq, wqt, H, DM, DK);
  transpose_cvt_kernel<<<(int)(nW / 256), 256, 0, stream>>>(Wk, wkt, H, DM, DK);
  transpose_cvt_kernel<<<(int)(nW / 256), 256, 0, stream>>>(Wv, wvt, H, DM, DK);
  transpose_cvt_kernel<<<(int)((size_t)DM * DM / 256), 256, 0, stream>>>(Wo, wot, 1, DM, DM);

  qkv_proj_kernel<<<dim3((B * S) / 64, H, 3), 128, 0, stream>>>(
      qb, kb, vb, wqt, wkt, wvt, Qp, Kp, Vt);

  attn_kernel<<<B * H * (S / 64), 128, 0, stream>>>(Qp, Kp, Vt, mlt);

  out_proj_kernel<<<dim3((B * S) / 64, DM / 64), 128, 0, stream>>>(
      mlt, wot, (float*)d_out);
}
